// AttentionLayer_56702158242250
// MI455X (gfx1250) — compile-verified, matrix operand order checked by reading, not run on hardware
//
#include <hip/hip_runtime.h>

// Problem constants (match reference): L=R=H=8, D=64, B=2, S=128.
#define NB   2
#define NS   128
#define NTOK (NB * NS)       // 256 tokens
#define NBLR (NB * 8 * 8)    // 128 (b,l,r) batch-heads
#define NROW 1024            // Sq*8 rows per batch-head (q*8+x), also Sk*8 cols (k*8+y)

typedef __attribute__((ext_vector_type(16))) _Float16 v16h;
typedef __attribute__((ext_vector_type(8)))  _Float16 v8h;
typedef __attribute__((ext_vector_type(8)))  float    v8f;
typedef __attribute__((ext_vector_type(4)))  int      v4i;

union FragA { v16h v; _Float16 h[16]; v8h h2[2]; };
union FragV { v8h  v; _Float16 h[8]; };
union AccF  { v8f  v; float f[8]; };

// gfx1250 async memory->LDS path (ASYNCcnt), guarded so either toolchain builds.
#if __has_builtin(__builtin_amdgcn_global_load_async_to_lds_b128) && \
    __has_builtin(__builtin_amdgcn_s_wait_asynccnt)
#define HAVE_ASYNC_LDS 1
typedef __attribute__((address_space(1))) v4i gv4i;   // global int4*
typedef __attribute__((address_space(3))) v4i lv4i;   // LDS int4*
#else
#define HAVE_ASYNC_LDS 0
#endif

static __device__ __forceinline__ v8h ld8h(const _Float16* p) {
    return *(const v8h*)p;
}

static __device__ __forceinline__ v8f wmma16(v16h a, v16h b, v8f c) {
    // D = A(16x32 f16) x B(32x16 f16) + C(16x16 f32)
    return __builtin_amdgcn_wmma_f32_16x16x32_f16(false, a, false, b, (short)0, c,
                                                  false, false);
}

// ---------------------------------------------------------------------------
// Kernel 0: convert the 8 weight matrices (64x64 f32) to f16, TRANSPOSED.
//   dst[i][c*64 + a] = w[i][a*64 + c]
// (wl needs wl^T as the WMMA A operand; wr needs column access -> wr^T too.)
// ---------------------------------------------------------------------------
struct WPtrs { const float* w[8]; };

__global__ __launch_bounds__(256) void wconv_kernel(WPtrs wp, _Float16* __restrict__ dst) {
    int mat = blockIdx.x;                    // 0..7
    const float* src = wp.w[mat];
    _Float16* d = dst + mat * 4096;
    for (int e = threadIdx.x; e < 4096; e += 256) {
        int c = e >> 6, a = e & 63;
        d[e] = (_Float16)src[a * 64 + c];
    }
}

// ---------------------------------------------------------------------------
// Kernel 1: fused bilinear projection for one input (q, k or v).
//   OUT = (wl^T @ X @ wr + bias) * scale, per token, via WMMA.
// mode 0: store head-separated row layout  dst[blr][s*8+x][e]      (Q, K)
// mode 1: store transposed                 dst[blr][e][s*8+y]      (V)
// ---------------------------------------------------------------------------
__global__ __launch_bounds__(128) void proj_qkv_kernel(
    const float* __restrict__ X,        // [NTOK][64][64] f32
    const _Float16* __restrict__ wlT,   // [64][64] = wl^T
    const _Float16* __restrict__ wrT,   // [64][64] = wr^T
    const float* __restrict__ bias,     // [64][64] f32
    _Float16* __restrict__ dst,
    float scale, int mode)
{
    __shared__ __align__(16) _Float16 XhT[64][64];   // X transposed, f16
    __shared__ __align__(16) _Float16 Th[64][64];    // T = wl^T @ X, f16

    const int tid  = threadIdx.x;
    const int lane = tid & 31;
    const int wave = tid >> 5;          // 0..3, owns row-strip tm = wave
    const int tok  = blockIdx.x;
    const int b    = tok >> 7;          // NS = 128
    const int s    = tok & 127;

    const float* Xp = X + (size_t)tok * 4096;

    // Load X (f32) -> XhT (f16, transposed) : 1024 float4 total
    for (int i = 0; i < 8; ++i) {
        int f4 = tid * 8 + i;                  // 0..1023
        int a  = f4 >> 4;                      // row of X
        int b0 = (f4 & 15) * 4;                // col base
        float4 q4 = *(const float4*)(Xp + a * 64 + b0);
        XhT[b0 + 0][a] = (_Float16)q4.x;
        XhT[b0 + 1][a] = (_Float16)q4.y;
        XhT[b0 + 2][a] = (_Float16)q4.z;
        XhT[b0 + 3][a] = (_Float16)q4.w;
    }
    __syncthreads();

    const int l15 = lane & 15;
    const int hi  = lane >> 4;          // 0 or 1
    const int o0  = hi ? 8 : 0;         // K-slot base per ISA A/B layout
    const int tm  = wave;

    // ---- T = wl^T @ X ----
    for (int tn = 0; tn < 4; ++tn) {
        AccF acc = {};
        for (int kk = 0; kk < 2; ++kk) {
            FragA A, Bf;
            const _Float16* ap = wlT + (tm * 16 + l15) * 64 + kk * 32 + o0;
            A.h2[0] = ld8h(ap);  A.h2[1] = ld8h(ap + 16);
            const _Float16* bp = &XhT[tn * 16 + l15][kk * 32 + o0];
            Bf.h2[0] = ld8h(bp); Bf.h2[1] = ld8h(bp + 16);
            acc.v = wmma16(A.v, Bf.v, acc.v);
        }
        #pragma unroll
        for (int j = 0; j < 8; ++j)
            Th[tm * 16 + hi * 8 + j][tn * 16 + l15] = (_Float16)acc.f[j];
    }
    __syncthreads();

    // ---- OUT = T @ wr ----
    for (int tn = 0; tn < 4; ++tn) {
        AccF acc = {};
        for (int kk = 0; kk < 2; ++kk) {
            FragA A, Bf;
            const _Float16* ap = &Th[tm * 16 + l15][kk * 32 + o0];
            A.h2[0] = ld8h(ap);  A.h2[1] = ld8h(ap + 16);
            const _Float16* bp = wrT + (tn * 16 + l15) * 64 + kk * 32 + o0;
            Bf.h2[0] = ld8h(bp); Bf.h2[1] = ld8h(bp + 16);
            acc.v = wmma16(A.v, Bf.v, acc.v);
        }
        // Epilogue: bias + scale, scatter by heads.
        const int d     = tn * 16 + l15;        // col of OUT = r*8+e
        const int cbase = tm * 16 + hi * 8;     // row base = l*8+x
        const int l = 2 * tm + hi;
        const int r = 2 * tn + (l15 >> 3);
        const int e = l15 & 7;
        const int blr = (b * 8 + l) * 8 + r;
        if (mode == 0) {
            size_t base = ((size_t)blr * NROW + (size_t)s * 8) * 8 + e;
            #pragma unroll
            for (int j = 0; j < 8; ++j) {       // j = x
                float o = (acc.f[j] + bias[(cbase + j) * 64 + d]) * scale;
                dst[base + (size_t)j * 8] = (_Float16)o;
            }
        } else {
            size_t base = ((size_t)blr * 8 + e) * NROW + (size_t)s * 8;
            FragV pk;
            #pragma unroll
            for (int j = 0; j < 8; ++j)         // j = y
                pk.h[j] = (_Float16)((acc.f[j] + bias[(cbase + j) * 64 + d]) * scale);
            *(v8h*)(dst + base) = pk.v;
        }
    }
}

// ---------------------------------------------------------------------------
// Kernel 2: attention per (b,l,r).  Scores K=8 padded into 16x16x32 WMMA,
// online softmax stats in registers (per-lane y = lane&7 after C-frag),
// merge via WMMA with V padded to N=16.  Output stored transposed f16.
// K/V staged to LDS with GLOBAL_LOAD_ASYNC_TO_LDS_B128 when available.
// ---------------------------------------------------------------------------
__global__ __launch_bounds__(256) void attn_kernel(
    const _Float16* __restrict__ Qh,    // [NBLR][1024][8]  (scaled by 1/64)
    const _Float16* __restrict__ Khg,   // [NBLR][1024][8]  (scaled by 1/8)
    const _Float16* __restrict__ Vtg,   // [NBLR][8][1024]  (scaled by 1/8)
    _Float16* __restrict__ Yt)          // [NTOK][64][64] transposed per token
{
    __shared__ __align__(16) _Float16 Kh[NROW * 8];
    __shared__ __align__(16) _Float16 Vt[8 * NROW];
    __shared__ __align__(16) _Float16 Wt[8][16 * 32];   // per-wave transpose tile

    const int tid  = threadIdx.x;
    const int lane = tid & 31;
    const int wave = tid >> 5;              // 8 waves
    const int blr  = blockIdx.x;
    const int bb   = blr >> 6;
    const int ll   = (blr >> 3) & 7;
    const int rr   = blr & 7;
    const size_t base = (size_t)blr * (NROW * 8);

    __builtin_prefetch(Qh + base, 0, 1);    // global_prefetch_b8

    // Stage K and V^T into LDS (16 KB each).
#if HAVE_ASYNC_LDS
    {
        gv4i* ks = (gv4i*)(const void*)(Khg + base);
        gv4i* vs = (gv4i*)(const void*)(Vtg + base);
        lv4i* kd = (lv4i*)(void*)Kh;
        lv4i* vd = (lv4i*)(void*)Vt;
        #pragma unroll
        for (int i = 0; i < 4; ++i) {
            int idx = i * 256 + tid;            // v4i (16B) elements
            __builtin_amdgcn_global_load_async_to_lds_b128(ks + idx, kd + idx, 0, 0);
            __builtin_amdgcn_global_load_async_to_lds_b128(vs + idx, vd + idx, 0, 0);
        }
        __builtin_amdgcn_s_wait_asynccnt(0);    // s_wait_asynccnt 0
    }
#else
    {
        const v8h* ks = (const v8h*)(Khg + base);
        const v8h* vs = (const v8h*)(Vtg + base);
        v8h* kd = (v8h*)Kh;
        v8h* vd = (v8h*)Vt;
        for (int i = tid; i < NROW; i += 256) { kd[i] = ks[i]; vd[i] = vs[i]; }
    }
#endif
    __syncthreads();

    const int l15 = lane & 15;
    const int hi  = lane >> 4;
    const int o0  = hi ? 8 : 0;
    const bool lo16 = (hi == 0);
    _Float16* Wtw = Wt[wave];

    for (int rt = wave; rt < 64; rt += 8) { // 16-row tiles, 8 per wave
        // Q fragment: only lanes 0-15 carry K=0..7 (rest of K padded to 0).
        FragA Qf = {};
        if (lo16) Qf.h2[0] = ld8h(Qh + base + (size_t)(rt * 16 + l15) * 8);

        // ---- Pass 1: scores + online max/sum.  Per lane: 8 rows, y = lane&7.
        float m[8], lsum[8];
        #pragma unroll
        for (int j = 0; j < 8; ++j) { m[j] = -1e30f; lsum[j] = 0.f; }

        for (int ct = 0; ct < 64; ++ct) {
            FragA Kf = {};
            if (lo16) Kf.h2[0] = ld8h(Kh + (size_t)(ct * 16 + l15) * 8);
            AccF c = {};
            c.v = wmma16(Qf.v, Kf.v, c.v);
            #pragma unroll
            for (int j = 0; j < 8; ++j) {
                float sv = c.f[j];
                float mn = fmaxf(m[j], sv);
                lsum[j] = lsum[j] * __expf(m[j] - mn) + __expf(sv - mn);
                m[j] = mn;
            }
        }
        // Merge the two k-parities (cols n and n^8 share y and rows).
        float invl[8];
        #pragma unroll
        for (int j = 0; j < 8; ++j) {
            float mo = __shfl_xor(m[j], 8, 32);
            float lo = __shfl_xor(lsum[j], 8, 32);
            float mn = fmaxf(m[j], mo);
            lsum[j]  = lsum[j] * __expf(m[j] - mn) + lo * __expf(mo - mn);
            m[j]     = mn;
            invl[j]  = 1.0f / lsum[j];
        }

        // ---- Pass 2: recompute scores, normalize, merge-GEMM with V.
        AccF acc = {};
        for (int kc = 0; kc < 32; ++kc) {       // K-chunks of 32 (k,y) columns
            #pragma unroll
            for (int hf = 0; hf < 2; ++hf) {
                int ct = kc * 2 + hf;
                FragA Kf = {};
                if (lo16) Kf.h2[0] = ld8h(Kh + (size_t)(ct * 16 + l15) * 8);
                AccF c = {};
                c.v = wmma16(Qf.v, Kf.v, c.v);
                #pragma unroll
                for (int j = 0; j < 8; ++j) {
                    float w = __expf(c.f[j] - m[j]) * invl[j];
                    Wtw[(hi * 8 + j) * 32 + hf * 16 + l15] = (_Float16)w;
                }
            }
            __threadfence_block();              // order DS stores before loads

            FragA Wf;                           // A-operand: normalized weights
            {
                const _Float16* wp = Wtw + l15 * 32 + o0;
                Wf.h2[0] = ld8h(wp); Wf.h2[1] = ld8h(wp + 16);
            }
            FragA Vf = {};                      // B-operand: V^T, N padded to 16
            if (l15 < 8) {
                const _Float16* vp = Vt + (size_t)l15 * NROW + kc * 32 + o0;
                Vf.h2[0] = ld8h(vp); Vf.h2[1] = ld8h(vp + 16);
            }
            acc.v = wmma16(Wf.v, Vf.v, acc.v);
            __threadfence_block();              // before next chunk overwrites Wt
        }

        // Store out tile transposed: Yt[tok][r*8+e][l*8+x], x = j contiguous.
        if (l15 < 8) {
            int q = rt * 2 + hi;
            int e = l15;
            FragV pk;
            #pragma unroll
            for (int j = 0; j < 8; ++j) pk.h[j] = (_Float16)acc.f[j];
            size_t addr = (((size_t)(bb * NS + q) * 64) + rr * 8 + e) * 64 + ll * 8;
            *(v8h*)(Yt + addr) = pk.v;
        }
    }
}

// ---------------------------------------------------------------------------
// Kernel 3: output projection  OUT = (wo_l^T @ Y @ wo_r + bo) / 8  -> f32.
// Y supplied already transposed+f16 (Yt), so it feeds B-fragments directly.
// ---------------------------------------------------------------------------
__global__ __launch_bounds__(128) void proj_out_kernel(
    const _Float16* __restrict__ Yt,    // [NTOK][64][64] (Y^T per token)
    const _Float16* __restrict__ wlT,
    const _Float16* __restrict__ wrT,
    const float* __restrict__ bias,
    float* __restrict__ out)            // [NTOK][64][64] f32
{
    __shared__ __align__(16) _Float16 Th[64][64];

    const int tid  = threadIdx.x;
    const int lane = tid & 31;
    const int wave = tid >> 5;
    const int tok  = blockIdx.x;
    const _Float16* Yp = Yt + (size_t)tok * 4096;

    const int l15 = lane & 15;
    const int hi  = lane >> 4;
    const int o0  = hi ? 8 : 0;
    const int tm  = wave;

    for (int tn = 0; tn < 4; ++tn) {
        AccF acc = {};
        for (int kk = 0; kk < 2; ++kk) {
            FragA A, Bf;
            const _Float16* ap = wlT + (tm * 16 + l15) * 64 + kk * 32 + o0;
            A.h2[0] = ld8h(ap);  A.h2[1] = ld8h(ap + 16);
            const _Float16* bp = Yp + (size_t)(tn * 16 + l15) * 64 + kk * 32 + o0;
            Bf.h2[0] = ld8h(bp); Bf.h2[1] = ld8h(bp + 16);
            acc.v = wmma16(A.v, Bf.v, acc.v);
        }
        #pragma unroll
        for (int j = 0; j < 8; ++j)
            Th[tm * 16 + hi * 8 + j][tn * 16 + l15] = (_Float16)acc.f[j];
    }
    __syncthreads();

    float* op = out + (size_t)tok * 4096;
    for (int tn = 0; tn < 4; ++tn) {
        AccF acc = {};
        for (int kk = 0; kk < 2; ++kk) {
            FragA A, Bf;
            const _Float16* ap = &Th[tm * 16 + l15][kk * 32 + o0];
            A.h2[0] = ld8h(ap);  A.h2[1] = ld8h(ap + 16);
            const _Float16* bp = wrT + (tn * 16 + l15) * 64 + kk * 32 + o0;
            Bf.h2[0] = ld8h(bp); Bf.h2[1] = ld8h(bp + 16);
            acc.v = wmma16(A.v, Bf.v, acc.v);
        }
        const int d     = tn * 16 + l15;
        const int cbase = tm * 16 + hi * 8;
        #pragma unroll
        for (int j = 0; j < 8; ++j)
            op[(cbase + j) * 64 + d] =
                (acc.f[j] + bias[(cbase + j) * 64 + d]) * 0.125f;
    }
}

// ---------------------------------------------------------------------------
extern "C" void kernel_launch(void* const* d_in, const int* in_sizes, int n_in,
                              void* d_out, int out_size, void* d_ws, size_t ws_size,
                              hipStream_t stream) {
    (void)in_sizes; (void)n_in; (void)out_size; (void)ws_size;

    const float* queries = (const float*)d_in[0];
    const float* keys    = (const float*)d_in[1];
    const float* values  = (const float*)d_in[2];
    const float* wq_l = (const float*)d_in[3],  *wq_r = (const float*)d_in[4],
               * bq   = (const float*)d_in[5];
    const float* wk_l = (const float*)d_in[6],  *wk_r = (const float*)d_in[7],
               * bk   = (const float*)d_in[8];
    const float* wv_l = (const float*)d_in[9],  *wv_r = (const float*)d_in[10],
               * bv   = (const float*)d_in[11];
    const float* wo_l = (const float*)d_in[12], *wo_r = (const float*)d_in[13],
               * bo   = (const float*)d_in[14];

    char* ws = (char*)d_ws;
    _Float16* Wh = (_Float16*)ws;                       // 8 * 4096 halfs (64 KB)
    _Float16* Qh = (_Float16*)(ws + (64 << 10));        // NBLR*8192 halfs (2 MB)
    _Float16* Kh = Qh + (size_t)NBLR * 8192;            // 2 MB
    _Float16* Vt = Kh + (size_t)NBLR * 8192;            // 2 MB
    _Float16* Yt = Vt + (size_t)NBLR * 8192;            // 2 MB

    WPtrs wp = {{wq_l, wq_r, wk_l, wk_r, wv_l, wv_r, wo_l, wo_r}};
    wconv_kernel<<<8, 256, 0, stream>>>(wp, Wh);

    // Q gets the extra 1/h of the sim-einsum folded in: scale = 1/64.
    proj_qkv_kernel<<<NTOK, 128, 0, stream>>>(queries, Wh + 0 * 4096, Wh + 1 * 4096,
                                              bq, Qh, 0.015625f, 0);
    proj_qkv_kernel<<<NTOK, 128, 0, stream>>>(keys,    Wh + 2 * 4096, Wh + 3 * 4096,
                                              bk, Kh, 0.125f, 0);
    proj_qkv_kernel<<<NTOK, 128, 0, stream>>>(values,  Wh + 4 * 4096, Wh + 5 * 4096,
                                              bv, Vt, 0.125f, 1);

    attn_kernel<<<NBLR, 256, 0, stream>>>(Qh, Kh, Vt, Yt);

    proj_out_kernel<<<NTOK, 128, 0, stream>>>(Yt, Wh + 6 * 4096, Wh + 7 * 4096,
                                              bo, (float*)d_out);
}
